// BahdanauAttentionMechanism_28346784153946
// MI455X (gfx1250) — matrix operand order repaired and checked by reading, not compiled, not run on hardware
//
#include <hip/hip_runtime.h>
#include <math.h>

// ---------------------------------------------------------------------------
// Bahdanau attention for MI455X (gfx1250, wave32).
// proj_v GEMM done with v_wmma_f32_16x16x32_bf16 using hi/lo fp32->2xbf16
// split (3 products: hh + hl + lh) for near-fp32 accuracy at bf16-pipe speed.
// Round 3: fix async-LDS builtin pointer types (v4i AS1 src / v4i AS3 dst).
// ---------------------------------------------------------------------------

typedef __attribute__((ext_vector_type(16))) __bf16 v16bf;
typedef __attribute__((ext_vector_type(8)))  float  v8f;
typedef __attribute__((ext_vector_type(4)))  int    v4i;

#define B_   32
#define T_   4096
#define D_   256
#define DQ_  512

// Workspace layout (bytes)
#define WS_PROJQ   0                              // 32*256 f32   = 32 KB
#define WS_SCORES  (32768)                        // 32*4096 f32  = 512 KB
#define WS_WVH     (32768 + 524288)               // 65536 bf16   = 128 KB
#define WS_WVL     (32768 + 524288 + 131072)      // 65536 bf16   = 128 KB

#define AS1 __attribute__((address_space(1)))
#define AS3 __attribute__((address_space(3)))
typedef AS1 v4i* gv4i_p;   // global-AS pointer to int4 (builtin src type)
typedef AS3 v4i* lv4i_p;   // LDS-AS pointer to int4    (builtin dst type)

#if __has_builtin(__builtin_amdgcn_global_load_async_to_lds_b128) && \
    __has_builtin(__builtin_amdgcn_s_wait_asynccnt)
#define USE_ASYNC_LDS 1
#else
#define USE_ASYNC_LDS 0
#endif

// ---------------------------------------------------------------------------
// proj_q[b][e] = sum_q query[b][q] * Wu[q][e]     (tiny GEMM, 32x512x256)
// ---------------------------------------------------------------------------
__global__ __launch_bounds__(256) void prep_projq(const float* __restrict__ query,
                                                  const float* __restrict__ Wu,
                                                  float* __restrict__ projq) {
    const int b = blockIdx.x;
    const int e = threadIdx.x;
    float acc = 0.f;
    #pragma unroll 8
    for (int q = 0; q < DQ_; ++q)
        acc += query[b * DQ_ + q] * Wu[q * D_ + e];   // Wu coalesced over e
    projq[b * D_ + e] = acc;
}

// ---------------------------------------------------------------------------
// Pack Wv (fp32 [K=256][N=256]) into bf16 hi/lo in WMMA B-operand order.
// Block index blk = nt*8 + ks covers B-tile (K = ks*32..+31, N = nt*16..+15).
// Entry e2 = lane*16 + elem; elems 0-7 hold K=k0+8*(lane>=16)+elem, elems
// 8-15 hold K=k0+16+8*(lane>=16)+elem-8; N = n0 + (lane&15).  (32x16 bf16 B layout, wave32)
// ---------------------------------------------------------------------------
__global__ __launch_bounds__(256) void prep_pack(const float* __restrict__ Wv,
                                                 __bf16* __restrict__ H,
                                                 __bf16* __restrict__ L) {
    const int blk = blockIdx.x;          // 0..127
    const int nt  = blk >> 3;
    const int ks  = blk & 7;
    for (int e2 = threadIdx.x; e2 < 512; e2 += 256) {
        const int lane = e2 >> 4;
        const int el   = e2 & 15;
        const int K = ks * 32 + ((lane < 16) ? 0 : 8) + ((el < 8) ? el : 8 + el);
        const int N = nt * 16 + (lane & 15);
        const float w = Wv[K * D_ + N];
        const __bf16 h = (__bf16)w;
        H[blk * 512 + e2] = h;
        L[blk * 512 + e2] = (__bf16)(w - (float)h);
    }
}

// ---------------------------------------------------------------------------
// Main WMMA kernel: scores[b][t] = sum_e tanh(proj_v[t][e]+projq[b][e])*Ws[e]
// 8 waves per block, one 16-row T-tile per wave. A cached in VGPRs (full K),
// Wv hi/lo streamed through 64 KB LDS in 4 N-groups of 4 N-tiles.
// ---------------------------------------------------------------------------
__global__ __launch_bounds__(256) void score_kernel(const float* __restrict__ values,
                                                    const float* __restrict__ projq,
                                                    const float* __restrict__ wscore,
                                                    const __bf16* __restrict__ WvH,
                                                    const __bf16* __restrict__ WvL,
                                                    float* __restrict__ scores) {
    // 4 n-tiles * 8 k-steps * 32 lanes * 16 halves = 16384 halves = 32 KB each
    __shared__ __attribute__((aligned(32))) __bf16 ldsH[16384];
    __shared__ __attribute__((aligned(32))) __bf16 ldsL[16384];

    const int tid  = threadIdx.x;
    const int wave = tid >> 5;
    const int lane = tid & 31;
    const int tile = blockIdx.x * 8 + wave;   // 0..8191 (= B * T/16)
    const int b    = tile >> 8;               // T/16 == 256 tiles per batch
    const int t0   = (tile & 255) << 4;

    // ---- Load A (16x256 fp32 rows of values) into bf16 hi/lo VGPRs ----
    // A 16x32 layout per k-step: lanes 0-15: M=lane, elems 0-7 = K k0..k0+7,
    // elems 8-15 = K k0+16..k0+23; lanes 16-31: M=lane-16, K offsets +8.
    const float* rowp =
        values + ((size_t)(b * T_ + t0 + (lane & 15))) * D_;
    const int koff = (lane >= 16) ? 8 : 0;

    v16bf aH[8], aL[8];
    #pragma unroll
    for (int ks = 0; ks < 8; ++ks) {
        const float* p = rowp + ks * 32 + koff;
        float4 c0 = *(const float4*)(p);
        float4 c1 = *(const float4*)(p + 4);
        float4 c2 = *(const float4*)(p + 16);
        float4 c3 = *(const float4*)(p + 20);
        float f[16] = {c0.x, c0.y, c0.z, c0.w, c1.x, c1.y, c1.z, c1.w,
                       c2.x, c2.y, c2.z, c2.w, c3.x, c3.y, c3.z, c3.w};
        #pragma unroll
        for (int j = 0; j < 16; ++j) {
            __bf16 h = (__bf16)f[j];
            aH[ks][j] = h;
            aL[ks][j] = (__bf16)(f[j] - (float)h);
        }
    }

    float accs[8];
    #pragma unroll
    for (int r = 0; r < 8; ++r) accs[r] = 0.f;

    for (int g = 0; g < 4; ++g) {
        __syncthreads();
        // cooperative fill: contiguous 32 KB hi + 32 KB lo for this n-group
        {
            const uint4* srcH = (const uint4*)(WvH + g * 16384) + tid;
            const uint4* srcL = (const uint4*)(WvL + g * 16384) + tid;
            uint4* dH = (uint4*)ldsH + tid;
            uint4* dL = (uint4*)ldsL + tid;
#if USE_ASYNC_LDS
            #pragma unroll
            for (int i = 0; i < 8; ++i) {            // 2048 uint4 / 256 thr
                __builtin_amdgcn_global_load_async_to_lds_b128(
                    (gv4i_p)(srcH + i * 256), (lv4i_p)(dH + i * 256), 0, 0);
                __builtin_amdgcn_global_load_async_to_lds_b128(
                    (gv4i_p)(srcL + i * 256), (lv4i_p)(dL + i * 256), 0, 0);
            }
            __builtin_amdgcn_s_wait_asynccnt(0);
#else
            #pragma unroll
            for (int i = 0; i < 8; ++i) {            // 2048 uint4 / 256 thr
                dH[i * 256] = srcH[i * 256];
                dL[i * 256] = srcL[i * 256];
            }
#endif
        }
        __syncthreads();

        #pragma unroll
        for (int ntl = 0; ntl < 4; ++ntl) {
            const int nt = g * 4 + ntl;
            // 3 independent accumulator chains: hh, hl, lh split products
            v8f cHH = {};
            v8f cHL = {};
            v8f cLH = {};
            #pragma unroll
            for (int ks = 0; ks < 8; ++ks) {
                const int base = ((ntl * 8 + ks) * 32 + lane) * 16;
                v16bf bH = *(const v16bf*)&ldsH[base];
                v16bf bL = *(const v16bf*)&ldsL[base];
                cHH = __builtin_amdgcn_wmma_f32_16x16x32_bf16(
                        false, aH[ks], false, bH, (short)0, cHH, false, false);
                cHL = __builtin_amdgcn_wmma_f32_16x16x32_bf16(
                        false, aH[ks], false, bL, (short)0, cHL, false, false);
                cLH = __builtin_amdgcn_wmma_f32_16x16x32_bf16(
                        false, aL[ks], false, bH, (short)0, cLH, false, false);
            }
            // epilogue: lane holds column n = nt*16 + (lane&15),
            // rows M = r + 8*(lane>=16) in c[r]
            const int n = nt * 16 + (lane & 15);
            const float pq  = projq[b * D_ + n];
            const float wsc = wscore[n];
            #pragma unroll
            for (int r = 0; r < 8; ++r) {
                const float pv = (cHH[r] + cHL[r]) + cLH[r];
                accs[r] += tanhf(pv + pq) * wsc;
            }
        }
    }

    // reduce each row's partial over its 16-lane half (xor 1,2,4,8)
    #pragma unroll
    for (int r = 0; r < 8; ++r) {
        float v = accs[r];
        v += __shfl_xor(v, 1);
        v += __shfl_xor(v, 2);
        v += __shfl_xor(v, 4);
        v += __shfl_xor(v, 8);
        accs[r] = v;
    }
    if ((lane & 15) == 0) {
        const int hf = lane >> 4;   // 0: rows 0-7, 1: rows 8-15
        #pragma unroll
        for (int r = 0; r < 8; ++r)
            scores[b * T_ + t0 + hf * 8 + r] = accs[r];
    }
}

// ---------------------------------------------------------------------------
// Softmax over T per batch; writes alignments, zeroes attention slots.
// ---------------------------------------------------------------------------
__global__ __launch_bounds__(256) void softmax_kernel(const float* __restrict__ scores,
                                                      float* __restrict__ out) {
    const int b   = blockIdx.x;
    const int tid = threadIdx.x;
    __shared__ float red[256];

    float v[16];
    float m = -3.4e38f;
    #pragma unroll
    for (int i = 0; i < 16; ++i) {
        v[i] = scores[b * T_ + tid + i * 256];
        m = fmaxf(m, v[i]);
    }
    red[tid] = m;
    __syncthreads();
    for (int s = 128; s > 0; s >>= 1) {
        if (tid < s) red[tid] = fmaxf(red[tid], red[tid + s]);
        __syncthreads();
    }
    m = red[0];
    __syncthreads();

    float sum = 0.f;
    #pragma unroll
    for (int i = 0; i < 16; ++i) {
        v[i] = expf(v[i] - m);
        sum += v[i];
    }
    red[tid] = sum;
    __syncthreads();
    for (int s = 128; s > 0; s >>= 1) {
        if (tid < s) red[tid] += red[tid + s];
        __syncthreads();
    }
    const float inv = 1.f / red[0];

    float* align = out + B_ * D_;            // alignments after attentions
    #pragma unroll
    for (int i = 0; i < 16; ++i)
        align[b * T_ + tid + i * 256] = v[i] * inv;

    out[b * D_ + tid] = 0.f;                 // zero attention accumulators
}

// ---------------------------------------------------------------------------
// attentions[b][d] += sum_{t in chunk} align[b][t] * values[b][t][d]
// grid (chunks=16, B); values reads are L2-resident second pass.
// ---------------------------------------------------------------------------
__global__ __launch_bounds__(256) void wsum_kernel(const float* __restrict__ values,
                                                   const float* __restrict__ align,
                                                   float* __restrict__ att) {
    const int ck = blockIdx.x;   // 0..15, 256 t each
    const int b  = blockIdx.y;
    const int d  = threadIdx.x;
    __shared__ float sa[256];
    const int t0 = ck * 256;
    sa[d] = align[b * T_ + t0 + d];
    __syncthreads();

    float acc = 0.f;
    const float* vp = values + ((size_t)b * T_ + t0) * D_ + d;
    #pragma unroll 8
    for (int i = 0; i < 256; ++i)
        acc += sa[i] * vp[(size_t)i * D_];
    atomicAdd(&att[b * D_ + d], acc);
}

// ---------------------------------------------------------------------------
extern "C" void kernel_launch(void* const* d_in, const int* in_sizes, int n_in,
                              void* d_out, int out_size, void* d_ws, size_t ws_size,
                              hipStream_t stream) {
    (void)in_sizes; (void)n_in; (void)out_size; (void)ws_size;
    const float* query  = (const float*)d_in[0];   // [32,512]
    const float* values = (const float*)d_in[1];   // [32,4096,256]
    const float* Wv     = (const float*)d_in[2];   // [256,256]
    const float* Wu     = (const float*)d_in[3];   // [512,256]
    const float* Wscore = (const float*)d_in[4];   // [256,1]

    float*  projq  = (float*)((char*)d_ws + WS_PROJQ);
    float*  scores = (float*)((char*)d_ws + WS_SCORES);
    __bf16* wvH    = (__bf16*)((char*)d_ws + WS_WVH);
    __bf16* wvL    = (__bf16*)((char*)d_ws + WS_WVL);

    float* out   = (float*)d_out;          // [0,8192): attentions
    float* align = out + B_ * D_;          // [8192, 8192+131072): alignments

    prep_projq<<<B_, 256, 0, stream>>>(query, Wu, projq);
    prep_pack<<<128, 256, 0, stream>>>(Wv, wvH, wvL);
    score_kernel<<<(B_ * (T_ / 16)) / 8, 256, 0, stream>>>(
        values, projq, Wscore, wvH, wvL, scores);
    softmax_kernel<<<B_, 256, 0, stream>>>(scores, out);
    wsum_kernel<<<dim3(16, B_), 256, 0, stream>>>(values, align, out);
}
